// AudioToPose_28295244546851
// MI455X (gfx1250) — compile-verified, matrix operand order checked by reading, not run on hardware
//
#include <hip/hip_runtime.h>
#include <hip/hip_bf16.h>

// ---------------- problem dimensions ----------------
#define BS    32
#define SEQ   512
#define AD    512
#define AF    256
#define PE    9
#define D3    73
#define H5    45          // 5*PE
#define NG    1024        // 4*AF (encoder gates)
#define DECG  180         // 4*H5
#define DECGP 192         // DECG padded to multiple of 64 for WMMA
#define DIN0  813         // 3*AF + H5  (decoder cell0 input width)
#define AWK   768         // 3*AF       (precomputable part of decoder input)
#define SM1   (SEQ-1)     // 511 decoder steps
#define MROWS (SEQ*BS)    // 16384
#define MDEC  (SM1*BS)    // 16352

typedef __bf16 bf16;
typedef __attribute__((ext_vector_type(16))) __bf16 v16bf;
typedef __attribute__((ext_vector_type(8)))  __bf16 v8bf;
typedef __attribute__((ext_vector_type(8)))  float  v8f;

__device__ __forceinline__ float sigf(float x) { return 1.f / (1.f + __expf(-x)); }

// Opaque zero in an SGPR: adding it to a pointer preserves the pointer's
// (global) address space for codegen, but makes the resulting addresses
// unknown to the optimizer each time it is materialized -- this stops the
// compiler from hoisting all loop-invariant WMMA fragments out of the time
// loop (256 VGPRs worth) and spilling them to scratch.
__device__ __forceinline__ size_t opaque_zero() {
    size_t z = 0;
    asm volatile("" : "+s"(z));
    return z;
}

// ==================================================================
// Pack W [N, K] (f32 row-major, row stride ldw) into WMMA B-fragment
// layout (bf16). For tile (kt, nt): lane holds 16 contiguous bf16:
//   pack[((kt*(Npad/16)+nt)*512) + lane*16 + e] = W[n*ldw + k]
//   n = nt*16 + (lane&15),  k = kt*32 + (e<8 ? e : e+8) + (lane>>4)*8
// Zero-pads n >= N.
// ==================================================================
__global__ void pack_b_kernel(const float* __restrict__ W, bf16* __restrict__ pack,
                              int N, int ldw, int K, int Npad) {
    int idx = blockIdx.x * blockDim.x + threadIdx.x;
    int total = (K >> 5) * (Npad >> 4) * 512;
    if (idx >= total) return;
    int e    = idx & 15;
    int lane = (idx >> 4) & 31;
    int tile = idx >> 9;
    int ntiles = Npad >> 4;
    int kt = tile / ntiles, nt = tile % ntiles;
    int half = lane >> 4;
    int n = nt * 16 + (lane & 15);
    int k = kt * 32 + ((e & 8) ? (e + 8) : e) + half * 8;
    float v = (n < N) ? W[(size_t)n * ldw + k] : 0.f;
    pack[idx] = (bf16)v;
}

// ==================================================================
// LayerNorm over audio rows -> bf16, relaid out to [S, BS, AD]
// One wave (32 lanes) per row of 512.
// ==================================================================
__global__ void ln_audio_kernel(const float* __restrict__ audio,
                                const float* __restrict__ g, const float* __restrict__ be,
                                bf16* __restrict__ out) {
    int wid  = (blockIdx.x * blockDim.x + threadIdx.x) >> 5;
    int lane = threadIdx.x & 31;
    if (wid >= MROWS) return;
    int s = wid / BS, b = wid % BS;
    const float* row = audio + ((size_t)b * SEQ + s) * AD;
    float v[16];
    float sum = 0.f;
#pragma unroll
    for (int i = 0; i < 16; ++i) { v[i] = row[lane + i * 32]; sum += v[i]; }
#pragma unroll
    for (int o = 16; o >= 1; o >>= 1) sum += __shfl_xor(sum, o, 32);
    float mean = sum * (1.f / AD);
    float sq = 0.f;
#pragma unroll
    for (int i = 0; i < 16; ++i) { float d = v[i] - mean; sq += d * d; }
#pragma unroll
    for (int o = 16; o >= 1; o >>= 1) sq += __shfl_xor(sq, o, 32);
    float rstd = rsqrtf(sq * (1.f / AD) + 1e-5f);
    bf16* orow = out + (size_t)wid * AD;
#pragma unroll
    for (int i = 0; i < 16; ++i) {
        int d = lane + i * 32;
        orow[d] = (bf16)((v[i] - mean) * rstd * g[d] + be[d]);
    }
}

// ==================================================================
// Big WMMA GEMM:  C[M, Npad] = act(A[M, K] @ packedB + bias)
// A: bf16 row-major, B: pre-packed fragments, C: f32 or bf16.
// Block = 256 threads (8 waves); each wave -> 16 rows x 64 cols;
// block covers 128 x 64. B k-panel staged through LDS (shared by waves).
// ==================================================================
template <typename OUT_T, int ACT>
__global__ __launch_bounds__(256) void
gemm_wmma_kernel(const bf16* __restrict__ A, const bf16* __restrict__ Bp,
                 const float* __restrict__ bias, OUT_T* __restrict__ C,
                 int M, int Npad, int K) {
    __shared__ bf16 ldsB[4 * 512];                       // K32 x N64 panel (fragment order)
    int tid  = threadIdx.x;
    int w    = tid >> 5;
    int lane = tid & 31;
    int half = lane >> 4, lrow = lane & 15;
    int mbase = blockIdx.y * 128 + w * 16;
    int nt0   = blockIdx.x * 4;                          // first 16-wide n tile
    int ktiles = K >> 5;
    int ntiles = Npad >> 4;
    v8f acc[4] = {};
    int arow = mbase + lrow; if (arow >= M) arow = M - 1;
    const bf16* arowp = A + (size_t)arow * K + half * 8;

    for (int kt = 0; kt < ktiles; ++kt) {
        // stage B panel (4 consecutive packed tiles = 2048 bf16) into LDS
        {
            const bf16* src = Bp + (((size_t)kt * ntiles + nt0) << 9);
            *((v8bf*)(ldsB + tid * 8)) = *((const v8bf*)(src + tid * 8));
        }
        __syncthreads();
        // A fragment straight from global (L2-resident activations)
        const bf16* ap = arowp + kt * 32;
        if (kt + 1 < ktiles) __builtin_prefetch(ap + 32, 0, 1);
        v8bf lo = *(const v8bf*)ap;
        v8bf hi = *(const v8bf*)(ap + 16);
        v16bf afrag;
#pragma unroll
        for (int i = 0; i < 8; ++i) { afrag[i] = lo[i]; afrag[i + 8] = hi[i]; }
#pragma unroll
        for (int j = 0; j < 4; ++j) {
            v16bf bfrag = *((const v16bf*)(ldsB + j * 512) + lane);
            acc[j] = __builtin_amdgcn_wmma_f32_16x16x32_bf16(
                false, afrag, false, bfrag, (short)0, acc[j], false, false);
        }
        __syncthreads();
    }
#pragma unroll
    for (int j = 0; j < 4; ++j) {
        int n = (nt0 + j) * 16 + lrow;
        float bv = bias ? bias[n] : 0.f;
#pragma unroll
        for (int r = 0; r < 8; ++r) {
            int m = mbase + r + half * 8;
            if (m < M) {
                float v = acc[j][r] + bv;
                if (ACT == 1) v = tanhf(v);
                C[(size_t)m * Npad + n] = (OUT_T)v;
            }
        }
    }
}

// ==================================================================
// Persistent bidirectional-LSTM recurrence. grid=2 (dir), block=1024
// (32 waves). h (bf16) + c (f32) + gate-exchange live in LDS.
// Per step:  gates = pre[t] + h @ Whh^T (WMMA)  -> LSTM elementwise.
// Whh fragments are re-read from L2 each k-step through an opaque SGPR
// offset (unroll 2) so codegen keeps global_load_b128 (not flat) and
// the compiler cannot hoist 256 VGPRs of invariant fragments to scratch.
// ==================================================================
__global__ __launch_bounds__(1024, 1) void
lstm_dir_kernel(const float* __restrict__ preF, const float* __restrict__ preB,
                const bf16* __restrict__ WhhF, const bf16* __restrict__ WhhB,
                bf16* __restrict__ xs_out) {
    extern __shared__ char smem[];
    bf16*  hls = (bf16*)smem;                                   // [BS*AF] bf16
    float* cls = (float*)(smem + BS * AF * sizeof(bf16));       // [BS*AF] f32
    float* gls = (float*)(smem + BS * AF * sizeof(bf16) + BS * AF * sizeof(float)); // [BS*NG]

    int dir = blockIdx.x;
    const float* pre = dir ? preB : preF;
    const bf16*  WhhFo = WhhF;   // keep original kernel-arg pointers for addrspace
    const bf16*  WhhBo = WhhB;

    int tid = threadIdx.x;
    int w = tid >> 5, lane = tid & 31, half = lane >> 4, lrow = lane & 15;
    int mi = w & 1, ng64 = w >> 1;            // 2 m-tiles x 16 n-groups of 64
    const int ktiles = AF / 32;               // 8
    const int ntiles = NG / 16;               // 64

    for (int i = tid; i < BS * AF; i += blockDim.x) { hls[i] = (bf16)0.f; cls[i] = 0.f; }
    __syncthreads();

    for (int step = 0; step < SEQ; ++step) {
        int t = dir ? (SEQ - 1 - step) : step;
        v8f acc[4] = {};
#pragma unroll 2
        for (int kt = 0; kt < ktiles; ++kt) {
            // opaque per-iteration offset: keeps loads in global addrspace,
            // prevents cross-step hoisting of the whole fragment set
            size_t oz = opaque_zero();
            const bf16* whhF_s = WhhFo + oz;
            const bf16* whhB_s = WhhBo + oz;
            const bf16* whh_s  = dir ? whhB_s : whhF_s;
            const bf16* ap = hls + (mi * 16 + lrow) * AF + kt * 32 + half * 8;
            v8bf lo = *(const v8bf*)ap;
            v8bf hi = *(const v8bf*)(ap + 16);
            v16bf afrag;
#pragma unroll
            for (int i = 0; i < 8; ++i) { afrag[i] = lo[i]; afrag[i + 8] = hi[i]; }
#pragma unroll
            for (int j = 0; j < 4; ++j) {
                int nt = ng64 * 4 + j;
                v16bf bfrag = *((const v16bf*)(whh_s + (((size_t)kt * ntiles + nt) << 9)) + lane);
                acc[j] = __builtin_amdgcn_wmma_f32_16x16x32_bf16(
                    false, afrag, false, bfrag, (short)0, acc[j], false, false);
            }
        }
#pragma unroll
        for (int j = 0; j < 4; ++j) {
            int n = ng64 * 64 + j * 16 + lrow;
#pragma unroll
            for (int r = 0; r < 8; ++r) {
                int m = mi * 16 + r + half * 8;
                gls[m * NG + n] = acc[j][r];
            }
        }
        __syncthreads();
        const float* prerow = pre + (size_t)t * BS * NG;
        for (int cell = tid; cell < BS * AF; cell += blockDim.x) {
            int b = cell >> 8, j = cell & (AF - 1);
            float gi = prerow[b * NG + j]           + gls[b * NG + j];
            float gf = prerow[b * NG + AF + j]      + gls[b * NG + AF + j];
            float gg = prerow[b * NG + 2 * AF + j]  + gls[b * NG + 2 * AF + j];
            float go = prerow[b * NG + 3 * AF + j]  + gls[b * NG + 3 * AF + j];
            float c = cls[cell];
            c = sigf(gf) * c + sigf(gi) * tanhf(gg);
            float h = sigf(go) * tanhf(c);
            cls[cell] = c;
            hls[cell] = (bf16)h;
            xs_out[((size_t)t * BS + b) * (2 * AF) + dir * AF + j] = (bf16)h;
        }
        __syncthreads();
    }
}

// ==================================================================
// Gather audio-embedding windows [i-1, i, i+1] -> decoder A matrix (bf16)
// ==================================================================
__global__ void gather_dec_kernel(const bf16* __restrict__ ae, bf16* __restrict__ adec) {
    int idx = blockIdx.x * blockDim.x + threadIdx.x;
    int total = MDEC * AWK;
    if (idx >= total) return;
    int d = idx % AWK;
    int rb = idx / AWK;
    int b = rb % BS, t = rb / BS;
    int i = t + 1;
    int wsel = d / AF, dd = d % AF;
    int src_i = (wsel == 0) ? (i - 1) : ((wsel == 1) ? i : ((i + 1 < SEQ) ? (i + 1) : (SEQ - 1)));
    adec[idx] = ae[((size_t)src_i * BS + b) * AF + dd];
}

// ==================================================================
// Init-pose path (tiny): LN(init) -> MLP -> init_pos -> tem; also writes
// ori_pos to output row 0 and the padded decoder bias db0.
// ==================================================================
__global__ void init_path_kernel(const float* __restrict__ init,
                                 const float* __restrict__ g, const float* __restrict__ b,
                                 const float* __restrict__ W1, const float* __restrict__ b1,
                                 const float* __restrict__ W2, const float* __restrict__ b2,
                                 const float* __restrict__ Wp, const float* __restrict__ bp,
                                 const float* __restrict__ b_ih0, const float* __restrict__ b_hh0,
                                 float* __restrict__ tem, float* __restrict__ db0p,
                                 float* __restrict__ out) {
    __shared__ float ln[BS * D3], t1[BS * D3], t2[BS * D3], ip[BS * PE];
    int tid = threadIdx.x, nthr = blockDim.x;
    for (int bb = tid; bb < BS; bb += nthr) {
        const float* r = init + bb * D3;
        float s = 0.f;
        for (int d = 0; d < D3; ++d) s += r[d];
        float m = s / D3;
        float v = 0.f;
        for (int d = 0; d < D3; ++d) { float dd = r[d] - m; v += dd * dd; }
        float rstd = rsqrtf(v / D3 + 1e-5f);
        for (int d = 0; d < D3; ++d) ln[bb * D3 + d] = (r[d] - m) * rstd * g[d] + b[d];
    }
    __syncthreads();
    for (int i = tid; i < BS * D3; i += nthr) {
        int bb = i / D3, o = i % D3;
        float acc = b1[o];
        for (int d = 0; d < D3; ++d) acc += ln[bb * D3 + d] * W1[o * D3 + d];
        t1[i] = tanhf(acc);
    }
    __syncthreads();
    for (int i = tid; i < BS * D3; i += nthr) {
        int bb = i / D3, o = i % D3;
        float acc = b2[o];
        for (int d = 0; d < D3; ++d) acc += t1[bb * D3 + d] * W2[o * D3 + d];
        t2[i] = acc;
    }
    __syncthreads();
    for (int i = tid; i < BS * PE; i += nthr) {
        int bb = i / PE, pp = i % PE;
        int src = (pp < 3) ? pp : ((pp < 6) ? (67 + pp - 3) : (70 + pp - 6));
        ip[i] = t2[bb * D3 + src];
        out[(size_t)bb * SEQ * PE + pp] = init[bb * D3 + src];   // ori_pos -> row 0
    }
    __syncthreads();
    for (int i = tid; i < BS * H5; i += nthr) {
        int bb = i / H5, o = i % H5;
        float acc = bp[o];
        for (int k = 0; k < PE; ++k) acc += ip[bb * PE + k] * Wp[o * PE + k];
        tem[i] = acc;
    }
    for (int i = tid; i < DECGP; i += nthr)
        db0p[i] = (i < DECG) ? (b_ih0[i] + b_hh0[i]) : 0.f;
}

// ==================================================================
// Decoder autoregressive scan: 2-layer LSTM (hidden 45) + output proj.
// One workgroup; all weights + state in LDS. Latency-bound by design.
// ==================================================================
__global__ __launch_bounds__(512, 1) void
dec_scan_kernel(const float* __restrict__ awpre, const float* __restrict__ tem,
                const float* __restrict__ Wih0, const float* __restrict__ Whh0,
                const float* __restrict__ Wih1, const float* __restrict__ Whh1,
                const float* __restrict__ b_ih1, const float* __restrict__ b_hh1,
                const float* __restrict__ outW, const float* __restrict__ outb,
                float* __restrict__ out) {
    extern __shared__ float sm[];
    float* wI0 = sm;                       // [DECG*H5] tail of Wih0 (tp part)
    float* wH0 = wI0 + DECG * H5;
    float* wI1 = wH0 + DECG * H5;
    float* wH1 = wI1 + DECG * H5;
    float* wO  = wH1 + DECG * H5;          // [PE*H5]
    float* dB1 = wO + PE * H5;             // [DECG]
    float* h0  = dB1 + DECG;               // [BS*H5] x5
    float* c0  = h0 + BS * H5;
    float* h1  = c0 + BS * H5;
    float* c1  = h1 + BS * H5;
    float* tp  = c1 + BS * H5;
    float* gb  = tp + BS * H5;             // [BS*DECG]

    int tid = threadIdx.x, nthr = blockDim.x;
    for (int i = tid; i < DECG * H5; i += nthr) {
        int g = i / H5, k = i % H5;
        wI0[i] = Wih0[(size_t)g * DIN0 + AWK + k];
        wH0[i] = Whh0[i];
        wI1[i] = Wih1[i];
        wH1[i] = Whh1[i];
    }
    for (int i = tid; i < PE * H5; i += nthr) wO[i] = outW[i];
    for (int i = tid; i < DECG; i += nthr) dB1[i] = b_ih1[i] + b_hh1[i];
    for (int i = tid; i < BS * H5; i += nthr) {
        h0[i] = 0.f; c0[i] = 0.f; h1[i] = 0.f; c1[i] = 0.f;
        tp[i] = tem[i];
    }
    __syncthreads();

    for (int t = 0; t < SM1; ++t) {
        const float* aw = awpre + (size_t)t * BS * DECGP;
        for (int i = tid; i < BS * DECG; i += nthr) {
            int b = i / DECG, g = i % DECG;
            float acc = aw[b * DECGP + g];
            const float* wi = wI0 + g * H5;
            const float* wh = wH0 + g * H5;
            const float* tb = tp + b * H5;
            const float* hb = h0 + b * H5;
            for (int k = 0; k < H5; ++k) acc += wi[k] * tb[k] + wh[k] * hb[k];
            gb[i] = acc;
        }
        __syncthreads();
        for (int i = tid; i < BS * H5; i += nthr) {
            int b = i / H5, j = i % H5;
            float gi = gb[b * DECG + j],          gf = gb[b * DECG + H5 + j];
            float gg = gb[b * DECG + 2 * H5 + j], go = gb[b * DECG + 3 * H5 + j];
            float c = c0[i];
            c = sigf(gf) * c + sigf(gi) * tanhf(gg);
            c0[i] = c;
            h0[i] = sigf(go) * tanhf(c);
        }
        __syncthreads();
        for (int i = tid; i < BS * DECG; i += nthr) {
            int b = i / DECG, g = i % DECG;
            float acc = dB1[g];
            const float* wi = wI1 + g * H5;
            const float* wh = wH1 + g * H5;
            const float* ha = h0 + b * H5;
            const float* hb = h1 + b * H5;
            for (int k = 0; k < H5; ++k) acc += wi[k] * ha[k] + wh[k] * hb[k];
            gb[i] = acc;
        }
        __syncthreads();
        for (int i = tid; i < BS * H5; i += nthr) {
            int b = i / H5, j = i % H5;
            float gi = gb[b * DECG + j],          gf = gb[b * DECG + H5 + j];
            float gg = gb[b * DECG + 2 * H5 + j], go = gb[b * DECG + 3 * H5 + j];
            float c = c1[i];
            c = sigf(gf) * c + sigf(gi) * tanhf(gg);
            c1[i] = c;
            float h = sigf(go) * tanhf(c);
            h1[i] = h;
            tp[i] = h;
        }
        __syncthreads();
        for (int i = tid; i < BS * PE; i += nthr) {
            int b = i / PE, pp = i % PE;
            float acc = outb[pp];
            for (int k = 0; k < H5; ++k) acc += wO[pp * H5 + k] * h1[b * H5 + k];
            out[((size_t)b * SEQ + (t + 1)) * PE + pp] = acc;
        }
        __syncthreads();
    }
}

// ==================================================================
// Host orchestration
// ==================================================================
static inline size_t pk_elems(int K, int Npad) { return (size_t)(K / 32) * (Npad / 16) * 512; }

extern "C" void kernel_launch(void* const* d_in, const int* in_sizes, int n_in,
                              void* d_out, int out_size, void* d_ws, size_t ws_size,
                              hipStream_t stream) {
    (void)in_sizes; (void)n_in; (void)out_size; (void)ws_size;
    const float* audio       = (const float*)d_in[0];
    const float* initv       = (const float*)d_in[1];
    const float* ln_a_g      = (const float*)d_in[4];
    const float* ln_a_b      = (const float*)d_in[5];
    const float* fusion_W    = (const float*)d_in[6];
    const float* fusion_b    = (const float*)d_in[7];
    const float* encW[16];
    for (int i = 0; i < 16; ++i) encW[i] = (const float*)d_in[8 + i];  // per layer: fW_ih, fW_hh, bW_ih, bW_hh
    const float* emb_audio_W = (const float*)d_in[24];
    const float* emb_audio_b = (const float*)d_in[25];
    const float* ln_i_g      = (const float*)d_in[26];
    const float* ln_i_b      = (const float*)d_in[27];
    const float* gi_W1       = (const float*)d_in[28];
    const float* gi_b1       = (const float*)d_in[29];
    const float* gi_W2       = (const float*)d_in[30];
    const float* gi_b2       = (const float*)d_in[31];
    const float* emb_pos_W   = (const float*)d_in[32];
    const float* emb_pos_b   = (const float*)d_in[33];
    const float* decWp[8];
    for (int i = 0; i < 8; ++i) decWp[i] = (const float*)d_in[34 + i]; // l0: W_ih,W_hh,b_ih,b_hh ; l1: same
    const float* out_W       = (const float*)d_in[42];
    const float* out_b       = (const float*)d_in[43];
    float* outp = (float*)d_out;

    // ---- workspace bump allocator ----
    char* wp = (char*)d_ws;
    auto alloc = [&](size_t bytes) -> char* {
        char* r = wp;
        wp += (bytes + 255) & ~(size_t)255;
        return r;
    };
    bf16* fusion_pk = (bf16*)alloc(pk_elems(AD, AD) * sizeof(bf16));
    bf16* enc_ih_pk[8];
    bf16* enc_hh_pk[8];
    for (int l = 0; l < 4; ++l) {
        enc_ih_pk[2 * l]     = (bf16*)alloc(pk_elems(AD, NG) * sizeof(bf16));   // forward W_ih
        enc_hh_pk[2 * l]     = (bf16*)alloc(pk_elems(AF, NG) * sizeof(bf16));   // forward W_hh
        enc_ih_pk[2 * l + 1] = (bf16*)alloc(pk_elems(AD, NG) * sizeof(bf16));   // backward W_ih
        enc_hh_pk[2 * l + 1] = (bf16*)alloc(pk_elems(AF, NG) * sizeof(bf16));   // backward W_hh
    }
    bf16* emb_pk  = (bf16*)alloc(pk_elems(AD, AF) * sizeof(bf16));
    bf16* dec0_pk = (bf16*)alloc(pk_elems(AWK, DECGP) * sizeof(bf16));

    bf16*  a_ln    = (bf16*)alloc((size_t)MROWS * AD * sizeof(bf16));
    bf16*  xsA     = (bf16*)alloc((size_t)MROWS * AD * sizeof(bf16));
    bf16*  xsB     = (bf16*)alloc((size_t)MROWS * AD * sizeof(bf16));
    float* gates_f = (float*)alloc((size_t)MROWS * NG * sizeof(float));
    float* gates_b = (float*)alloc((size_t)MROWS * NG * sizeof(float));
    bf16*  ae      = (bf16*)alloc((size_t)MROWS * AF * sizeof(bf16));
    bf16*  adec    = (bf16*)alloc((size_t)MDEC * AWK * sizeof(bf16));
    float* awpre   = (float*)alloc((size_t)MDEC * DECGP * sizeof(float));
    float* tem     = (float*)alloc((size_t)BS * H5 * sizeof(float));
    float* db0p    = (float*)alloc((size_t)DECGP * sizeof(float));

    auto packN = [&](const float* W, bf16* dst, int N, int ldw, int K, int Npad) {
        int total = (K / 32) * (Npad / 16) * 512;
        pack_b_kernel<<<(total + 255) / 256, 256, 0, stream>>>(W, dst, N, ldw, K, Npad);
    };

    // ---- weight packing (bf16 fragment layout) ----
    packN(fusion_W, fusion_pk, AD, AD, AD, AD);
    for (int l = 0; l < 4; ++l) {
        packN(encW[4 * l + 0], enc_ih_pk[2 * l],     NG, AD, AD, NG);
        packN(encW[4 * l + 1], enc_hh_pk[2 * l],     NG, AF, AF, NG);
        packN(encW[4 * l + 2], enc_ih_pk[2 * l + 1], NG, AD, AD, NG);
        packN(encW[4 * l + 3], enc_hh_pk[2 * l + 1], NG, AF, AF, NG);
    }
    packN(emb_audio_W, emb_pk, AF, AD, AD, AF);
    packN(decWp[0], dec0_pk, DECG, DIN0, AWK, DECGP);   // decoder cell0, first 768 input cols

    // ---- LayerNorm + layout transform of audio ----
    ln_audio_kernel<<<(MROWS * 32) / 256, 256, 0, stream>>>(audio, ln_a_g, ln_a_b, a_ln);

    // ---- fusion: xs0 = tanh(LN(audio) @ fusion_W^T + b)  (bf16 out) ----
    gemm_wmma_kernel<bf16, 1><<<dim3(AD / 64, MROWS / 128), 256, 0, stream>>>(
        a_ln, fusion_pk, fusion_b, xsA, MROWS, AD, AD);

    // ---- 4-layer bidirectional LSTM encoder ----
    const size_t lstm_smem = (size_t)BS * AF * sizeof(bf16) + (size_t)BS * AF * sizeof(float)
                           + (size_t)BS * NG * sizeof(float);  // 176 KB
    bf16* xs_cur = xsA;
    bf16* xs_nxt = xsB;
    for (int l = 0; l < 4; ++l) {
        gemm_wmma_kernel<float, 0><<<dim3(NG / 64, MROWS / 128), 256, 0, stream>>>(
            xs_cur, enc_ih_pk[2 * l],     nullptr, gates_f, MROWS, NG, AD);
        gemm_wmma_kernel<float, 0><<<dim3(NG / 64, MROWS / 128), 256, 0, stream>>>(
            xs_cur, enc_ih_pk[2 * l + 1], nullptr, gates_b, MROWS, NG, AD);
        lstm_dir_kernel<<<2, 1024, lstm_smem, stream>>>(
            gates_f, gates_b, enc_hh_pk[2 * l], enc_hh_pk[2 * l + 1], xs_nxt);
        bf16* t = xs_cur; xs_cur = xs_nxt; xs_nxt = t;
    }

    // ---- audio embedding: ae = xs @ emb_W^T + b  (bf16 out) ----
    gemm_wmma_kernel<bf16, 0><<<dim3(AF / 64, MROWS / 128), 256, 0, stream>>>(
        xs_cur, emb_pk, emb_audio_b, ae, MROWS, AF, AD);

    // ---- decoder precompute: gather windows, init path, aw @ W_ih0[:, :768]^T ----
    {
        int total = MDEC * AWK;
        gather_dec_kernel<<<(total + 255) / 256, 256, 0, stream>>>(ae, adec);
    }
    init_path_kernel<<<1, 256, 0, stream>>>(initv, ln_i_g, ln_i_b, gi_W1, gi_b1, gi_W2, gi_b2,
                                            emb_pos_W, emb_pos_b, decWp[2], decWp[3],
                                            tem, db0p, outp);
    gemm_wmma_kernel<float, 0><<<dim3(DECGP / 64, (MDEC + 127) / 128), 256, 0, stream>>>(
        adec, dec0_pk, db0p, awpre, MDEC, DECGP, AWK);

    // ---- autoregressive decoder scan ----
    const size_t dec_smem = (size_t)(4 * DECG * H5 + PE * H5 + DECG + 5 * BS * H5 + BS * DECG)
                          * sizeof(float);  // ~184 KB
    dec_scan_kernel<<<1, 512, dec_smem, stream>>>(
        awpre, tem, decWp[0], decWp[1], decWp[4], decWp[5], decWp[6], decWp[7],
        out_W, out_b, outp);
}
